// CrossAttention_89962384982090
// MI455X (gfx1250) — matrix operand order repaired and checked by reading, not compiled, not run on hardware
//
#include <hip/hip_runtime.h>

// ---------------------------------------------------------------------------
// CDNA5 (gfx1250) fused cross-attention, bf16 WMMA (v_wmma_f32_16x16x32_bf16)
// B=4, H=8, D=64, N=M=2048, C=INNER=512
// ---------------------------------------------------------------------------

typedef __attribute__((ext_vector_type(16))) __bf16        v16bf;
typedef __attribute__((ext_vector_type(8)))  float         v8f;
typedef __attribute__((ext_vector_type(4)))  unsigned int  u32x4;

union FragB {
  v16bf  v;
  __bf16 e[16];
  u32x4  q[2];
};

__device__ __forceinline__ v8f wmma_bf16(v16bf a, v16bf b, v8f c) {
  return __builtin_amdgcn_wmma_f32_16x16x32_bf16(
      /*neg_a=*/false, a, /*neg_b=*/false, b,
      /*c_mod=*/(short)0, c, /*reuse_a=*/false, /*reuse_b=*/false);
}

// ---------------------------------------------------------------------------
// One-time weight conversion: all four 512x512 fp32 weights -> bf16 row-major.
// ---------------------------------------------------------------------------
__global__ __launch_bounds__(256) void cvt_w_kernel(
    const float* __restrict__ W0, const float* __restrict__ W1,
    const float* __restrict__ W2, const float* __restrict__ W3,
    __bf16* __restrict__ Wb) {
  const int SZ = 512 * 512;
  int t   = blockIdx.x * 256 + threadIdx.x;  // 262144 threads, 4 elems each
  int m   = t >> 16;
  int off = (t & 65535) << 2;
  const float* src = (m == 0) ? W0 : (m == 1) ? W1 : (m == 2) ? W2 : W3;
  float4 v = *(const float4*)(src + off);
  union { __bf16 e[4]; unsigned long long u; } p;
  p.e[0] = (__bf16)v.x; p.e[1] = (__bf16)v.y;
  p.e[2] = (__bf16)v.z; p.e[3] = (__bf16)v.w;
  *(unsigned long long*)(Wb + (size_t)m * SZ + off) = p.u;
}

// ---------------------------------------------------------------------------
// Projection: Out[o,s] = sum_c W[o,c] * X[b,c,s], bf16 output.
// Block = 8 waves covering all 512 o-rows of one 32-column slab; the 32x32
// x-tile is staged in LDS pre-swizzled into B-fragment order.
// TRANSPOSED=1 -> Out [b,h,s,64] (Q^T/K^T); 0 -> Out [b,h,64,s] (V)
// ---------------------------------------------------------------------------
template <int TRANSPOSED>
__global__ __launch_bounds__(256) void proj_kernel(
    const __bf16* __restrict__ Wb,   // [512,512] bf16 row-major
    const float* __restrict__ X,     // [B,512,S]
    __bf16* __restrict__ Out, int S) {
  const int C = 512;
  __shared__ __align__(32) __bf16 xs[2][512];  // two B-fragments, 1KB each

  int tid  = threadIdx.x;
  int wv   = tid >> 5;
  int lane = tid & 31;
  int lq = lane & 15, lh = lane >> 4;

  int slabs = S >> 5;
  int b  = blockIdx.x / slabs;
  int st = (blockIdx.x - b * slabs) << 5;
  int ob = wv << 6;

  int cc = tid >> 3;            // 0..31 (c offset within tile)
  int ss = (tid & 7) << 2;      // 0..28 (s offset within slab)
  const float* xg = X + ((size_t)b * C + cc) * S + st + ss;
  __bf16* dst = &xs[ss >> 4][((ss & 15) + 16 * ((cc >> 3) & 1)) * 16 + (cc & 7) + 8 * (cc >> 4)];

  v8f acc[8] = {};

  for (int kb = 0; kb < C; kb += 32) {
    float4 xv = *(const float4*)(xg + (size_t)kb * S);
    __syncthreads();
    dst[0]  = (__bf16)xv.x;
    dst[16] = (__bf16)xv.y;
    dst[32] = (__bf16)xv.z;
    dst[48] = (__bf16)xv.w;
    __syncthreads();

    v16bf bf0 = *(const v16bf*)(&xs[0][lane * 16]);
    v16bf bf1 = *(const v16bf*)(&xs[1][lane * 16]);

    const __bf16* wrow = Wb + (size_t)(ob + lq) * C + kb + 8 * lh;
    #pragma unroll
    for (int t = 0; t < 4; ++t) {
      FragB a;
      const __bf16* wp = wrow + (size_t)(16 * t) * C;
      a.q[0] = *(const u32x4*)(wp);
      a.q[1] = *(const u32x4*)(wp + 16);
      acc[2 * t]     = wmma_bf16(a.v, bf0, acc[2 * t]);
      acc[2 * t + 1] = wmma_bf16(a.v, bf1, acc[2 * t + 1]);
    }
  }

  #pragma unroll
  for (int t = 0; t < 4; ++t) {
    #pragma unroll
    for (int j = 0; j < 2; ++j) {
      v8f acv = acc[2 * t + j];
      int s = st + 16 * j + lq;
      #pragma unroll
      for (int r = 0; r < 8; ++r) {
        int o  = ob + 16 * t + r + 8 * lh;
        int hh = o >> 6, dd = o & 63;
        size_t idx = TRANSPOSED
            ? (((size_t)b * 8 + hh) * S + s) * 64 + dd
            : (((size_t)b * 8 + hh) * 64 + dd) * S + s;
        Out[idx] = (__bf16)acv[r];
      }
    }
  }
}

// ---------------------------------------------------------------------------
// Fused attention (flash style). Block = 8 waves sharing one (b,h) and a
// 128-query slab; each 32-key K/V tile is staged once in LDS, pre-swizzled
// into WMMA fragment order (4 K A-frags, 4 V B-frags, 1088B stride to spread
// staging-write banks). All 8 waves consume identical conflict-free 32B
// ds_loads -> 8x less global K/V traffic.
// ---------------------------------------------------------------------------
__global__ __launch_bounds__(256) void attn_kernel(
    const __bf16* __restrict__ Qt, const __bf16* __restrict__ Kt,
    const __bf16* __restrict__ Vv, __bf16* __restrict__ Ot) {
  const int N = 2048, M = 2048;
  const int FS = 544;                         // frag stride in halves (1088B)
  __shared__ __align__(32) __bf16 kfr[4 * FS];
  __shared__ __align__(32) __bf16 vfr[4 * FS];

  int tid  = threadIdx.x;
  int wv   = tid >> 5;
  int lane = tid & 31;
  int lq = lane & 15, lh = lane >> 4;

  int slabs = N >> 7;                         // 16 slabs of 128 queries
  int bh    = blockIdx.x / slabs;
  int nt    = ((blockIdx.x - bh * slabs) << 7) + (wv << 4);
  int b = bh >> 3, h = bh & 7;

  const __bf16* qb    = Qt + (size_t)bh * N * 64;
  const __bf16* kbase = Kt + (size_t)bh * M * 64;
  const __bf16* vbase = Vv + (size_t)bh * 64 * M;

  // staging roles (one 16B chunk per thread per tile)
  // K: key k = tid>>3, piece j = tid&7 : dd = 8j = 32c + 16lh + 8r
  int sk_k = tid >> 3, sk_j = tid & 7;
  const __bf16* kg = kbase + (size_t)sk_k * 64 + sk_j * 8;
  __bf16* kdst = kfr + ((sk_k >> 4) * 2 + (sk_j >> 2)) * FS
                     + ((sk_k & 15) + 16 * ((sk_j >> 1) & 1)) * 16
                     + (sk_j & 1) * 8;
  // V: dd = tid>>2, piece j16 = tid&3 : key off = 8*j16 = 8lh + 16r
  int sv_d = tid >> 2, sv_j = tid & 3;
  const __bf16* vg = vbase + (size_t)sv_d * M + sv_j * 8;
  __bf16* vdst = vfr + (sv_d >> 4) * FS
                     + ((sv_d & 15) + 16 * (sv_j & 1)) * 16
                     + (sv_j >> 1) * 8;

  // Q B-fragments, hoisted: single 32B loads
  v16bf q0 = *(const v16bf*)(qb + (size_t)(nt + lq) * 64 + 16 * lh);
  v16bf q1 = *(const v16bf*)(qb + (size_t)(nt + lq) * 64 + 32 + 16 * lh);

  const float SC = 0.125f * 1.44269504088896f;  // d^-0.5 * log2(e)
  float mrun = -3.0e38f, lrun = 0.0f;
  v8f o0 = {}, o1 = {}, o2 = {}, o3 = {};

  for (int mt = 0; mt < M; mt += 32) {
    u32x4 kc = *(const u32x4*)(kg + (size_t)mt * 64);
    u32x4 vc = *(const u32x4*)(vg + mt);
    __builtin_prefetch((const void*)(kg + (size_t)(mt + 32) * 64), 0, 0);
    __builtin_prefetch((const void*)(vg + mt + 32), 0, 0);
    __syncthreads();
    *(u32x4*)kdst = kc;
    *(u32x4*)vdst = vc;
    __syncthreads();

    // ---- S^T tiles from LDS fragments ----
    v8f s0 = {}, s1 = {};
    s0 = wmma_bf16(*(const v16bf*)(kfr + 0 * FS + lane * 16), q0, s0);
    s0 = wmma_bf16(*(const v16bf*)(kfr + 1 * FS + lane * 16), q1, s0);
    s1 = wmma_bf16(*(const v16bf*)(kfr + 2 * FS + lane * 16), q0, s1);
    s1 = wmma_bf16(*(const v16bf*)(kfr + 3 * FS + lane * 16), q1, s1);

    // ---- online softmax (exp2 domain), per query column ----
    float mloc = -3.0e38f;
    #pragma unroll
    for (int r = 0; r < 8; ++r) { mloc = fmaxf(mloc, s0[r]); mloc = fmaxf(mloc, s1[r]); }
    mloc = fmaxf(mloc, __shfl_xor(mloc, 16, 32));
    float mnew = fmaxf(mrun, mloc * SC);
    float fsc  = __builtin_amdgcn_exp2f(mrun - mnew);
    mrun = mnew;

    FragB pa;
    float psum = 0.0f;
    #pragma unroll
    for (int r = 0; r < 8; ++r) {
      float e0 = __builtin_amdgcn_exp2f(s0[r] * SC - mnew);
      float e1 = __builtin_amdgcn_exp2f(s1[r] * SC - mnew);
      psum += e0 + e1;
      pa.e[r]     = (__bf16)e0;
      pa.e[8 + r] = (__bf16)e1;
    }
    psum += __shfl_xor(psum, 16, 32);
    lrun = lrun * fsc + psum;

    #pragma unroll
    for (int r = 0; r < 8; ++r) {
      float fr = __shfl(fsc, r + 8 * lh, 32);
      o0[r] *= fr; o1[r] *= fr; o2[r] *= fr; o3[r] *= fr;
    }

    // ---- O += P * V^T from LDS fragments ----
    o0 = wmma_bf16(pa.v, *(const v16bf*)(vfr + 0 * FS + lane * 16), o0);
    o1 = wmma_bf16(pa.v, *(const v16bf*)(vfr + 1 * FS + lane * 16), o1);
    o2 = wmma_bf16(pa.v, *(const v16bf*)(vfr + 2 * FS + lane * 16), o2);
    o3 = wmma_bf16(pa.v, *(const v16bf*)(vfr + 3 * FS + lane * 16), o3);
  }

  #pragma unroll
  for (int r = 0; r < 8; ++r) {
    float inv = __builtin_amdgcn_rcpf(__shfl(lrun, r + 8 * lh, 32));
    size_t off = ((size_t)b * N + (nt + r + 8 * lh)) * 512 + (size_t)h * 64 + lq;
    Ot[off +  0] = (__bf16)(o0[r] * inv);
    Ot[off + 16] = (__bf16)(o1[r] * inv);
    Ot[off + 32] = (__bf16)(o2[r] * inv);
    Ot[off + 48] = (__bf16)(o3[r] * inv);
  }
}

// ---------------------------------------------------------------------------
// Output projection: Y[b,o,n] = sum_c Wo[o,c] * Ot[b,n,c] + bo[o]  (fp32 out)
// ---------------------------------------------------------------------------
__global__ __launch_bounds__(256) void out_proj_kernel(
    const __bf16* __restrict__ Wob, const __bf16* __restrict__ Ot,
    const float* __restrict__ bo, float* __restrict__ Y) {
  const int C = 512, N = 2048;
  int wid  = (blockIdx.x * blockDim.x + threadIdx.x) >> 5;
  int lane = threadIdx.x & 31;
  int lq = lane & 15, lh = lane >> 4;
  int tilesN = N >> 4;
  int perB   = 8 * tilesN;
  int b   = wid / perB;
  int rem = wid - b * perB;
  int tb  = rem / tilesN;
  int ob  = tb << 6;
  int nt  = (rem - tb * tilesN) << 4;

  const __bf16* orow = Ot + ((size_t)b * N + nt + lq) * 512;
  v8f acc[4] = {};

  for (int kb = 0; kb < C; kb += 32) {
    v16bf bfrag = *(const v16bf*)(orow + kb + 16 * lh);
    const __bf16* wrow = Wob + (size_t)(ob + lq) * C + kb + 16 * lh;
    #pragma unroll
    for (int t = 0; t < 4; ++t) {
      FragB a;
      const __bf16* wp = wrow + (size_t)(16 * t) * C;
      a.q[0] = *(const u32x4*)(wp);
      a.q[1] = *(const u32x4*)(wp + 8);
      acc[t] = wmma_bf16(a.v, bfrag, acc[t]);
    }
  }

  int n = nt + lq;
  #pragma unroll
  for (int t = 0; t < 4; ++t) {
    #pragma unroll
    for (int r = 0; r < 8; ++r) {
      int o = ob + 16 * t + r + 8 * lh;
      Y[((size_t)b * 512 + o) * N + n] = acc[t][r] + bo[o];
    }
  }
}

// ---------------------------------------------------------------------------
extern "C" void kernel_launch(void* const* d_in, const int* in_sizes, int n_in,
                              void* d_out, int out_size, void* d_ws, size_t ws_size,
                              hipStream_t stream) {
  const float* x   = (const float*)d_in[0];
  const float* ctx = (const float*)d_in[1];
  const float* Wq  = (const float*)d_in[2];
  const float* Wk  = (const float*)d_in[3];
  const float* Wv  = (const float*)d_in[4];
  const float* Wo  = (const float*)d_in[5];
  const float* bo  = (const float*)d_in[6];

  char* ws = (char*)d_ws;
  const int    WSZ  = 512 * 512;                      // elems per weight matrix
  const size_t WSEG = (size_t)4 * WSZ * 2;            // 2 MiB bf16 weights
  const size_t SEG  = (size_t)4 * 8 * 2048 * 64 * 2;  // 8 MiB per tensor, bf16
  __bf16* Wb = (__bf16*)(ws);                  // [4][512][512]
  __bf16* Qt = (__bf16*)(ws + WSEG);           // [B,H,N,64]
  __bf16* Kt = (__bf16*)(ws + WSEG + SEG);     // [B,H,M,64]
  __bf16* Vv = (__bf16*)(ws + WSEG + 2 * SEG); // [B,H,64,M]
  __bf16* Ot = (__bf16*)(ws + WSEG + 3 * SEG); // [B,N,512]

  dim3 blk(256);
  cvt_w_kernel<<<1024, blk, 0, stream>>>(Wq, Wk, Wv, Wo, Wb);
  proj_kernel<1><<<256, blk, 0, stream>>>(Wb,           x,   Qt, 2048);
  proj_kernel<1><<<256, blk, 0, stream>>>(Wb + WSZ,     ctx, Kt, 2048);
  proj_kernel<0><<<256, blk, 0, stream>>>(Wb + 2 * WSZ, ctx, Vv, 2048);
  // 32 (b,h) x 16 slabs = 512 blocks, 8 waves each (one 16-query tile per wave)
  attn_kernel<<<512, blk, 0, stream>>>(Qt, Kt, Vv, Ot);
  out_proj_kernel<<<512, blk, 0, stream>>>(Wb + 3 * WSZ, Ot, bo, (float*)d_out);
}
